// TransformerInferenceNetwork_3470333575730
// MI455X (gfx1250) — compile-verified, matrix operand order checked by reading, not run on hardware
//
#include <hip/hip_runtime.h>

#define DIM 128
#define NNODES 4096
#define NEDGES 8064

typedef __attribute__((ext_vector_type(16))) _Float16 v16h;
typedef __attribute__((ext_vector_type(8)))  float    v8f;
typedef int b128_t __attribute__((vector_size(16)));   // async-copy payload type

#if __has_builtin(__builtin_amdgcn_global_load_async_to_lds_b128) && \
    __has_builtin(__builtin_amdgcn_s_wait_asynccnt)
#define USE_ASYNC_LDS 1
#else
#define USE_ASYNC_LDS 0
#endif

__device__ __forceinline__ int kidx_of(int h, int hi) {
  // 16-bit A/B fragment layout: lanes 0-15 hold K = {0..7,16..23}, lanes 16-31 K = {8..15,24..31}
  return ((h >> 3) << 4) + (hi << 3) + (h & 7);
}

__device__ __forceinline__ v8f wmma_f16(v16h a, v16h b, v8f c) {
  return __builtin_amdgcn_wmma_f32_16x16x32_f16(false, a, false, b, (short)0, c,
                                                false, false);
}

// ---------------------------------------------------------------------------
// Pack fp32 row-major [R x (KC*32)] into 16x32 f16 A-fragments (also serves as
// B^T fragments: B[k][n] = M[n][k]).  frag f: rb = f/KC, kc = f%KC.
// ---------------------------------------------------------------------------
__global__ void pack_a_f32(const float* __restrict__ M, _Float16* __restrict__ dst,
                           int KC, int nfrags) {
  int gid = blockIdx.x * blockDim.x + threadIdx.x;
  int f = gid >> 5;
  if (f >= nfrags) return;
  int lane = gid & 31;
  int hi = lane >> 4, ml = lane & 15;
  int rb = f / KC, kc = f % KC;
  int Kcols = KC * 32;
  const float* src = M + (size_t)(rb * 16 + ml) * Kcols + kc * 32;
  v16h vals;
#pragma unroll
  for (int h = 0; h < 16; ++h) vals[h] = (_Float16)src[kidx_of(h, hi)];
  *(v16h*)(dst + ((size_t)f * 32 + lane) * 16) = vals;
}

// ---------------------------------------------------------------------------
// Pack fp32 row-major [ (KCtot*32) x Ncols ] into 32x16 f16 B-fragments,
// B[k][n] = M[k][n].  kc-major frag order: f = kc*NB + nb  (matches flash PV).
// ---------------------------------------------------------------------------
__global__ void pack_bN_f32(const float* __restrict__ M, _Float16* __restrict__ dst,
                            int NB, int Ncols, int nfrags) {
  int gid = blockIdx.x * blockDim.x + threadIdx.x;
  int f = gid >> 5;
  if (f >= nfrags) return;
  int lane = gid & 31;
  int hi = lane >> 4, nl = lane & 15;
  int kc = f / NB, nb = f % NB;
  v16h vals;
#pragma unroll
  for (int h = 0; h < 16; ++h)
    vals[h] = (_Float16)M[(size_t)(kc * 32 + kidx_of(h, hi)) * Ncols + nb * 16 + nl];
  *(v16h*)(dst + ((size_t)f * 32 + lane) * 16) = vals;
}

// ---------------------------------------------------------------------------
// Generic WMMA GEMM: C[M x N] = A[M x K] * B[K x N] (+bias) (relu) (+residual)
// A-frags: f = mb*KC + kc ; B-frags: f = nb*KC + kc. One wave per 16x16 tile.
// ---------------------------------------------------------------------------
__global__ void gemm16(const _Float16* __restrict__ A, const _Float16* __restrict__ B,
                       const float* __restrict__ bias, const float* __restrict__ residual,
                       float* __restrict__ Cout, int Mtiles, int Ntiles, int KC,
                       int do_relu) {
  int gid = blockIdx.x * blockDim.x + threadIdx.x;
  int w = gid >> 5;
  if (w >= Mtiles * Ntiles) return;
  int lane = gid & 31, hi = lane >> 4, nl = lane & 15;
  int mb = w / Ntiles, nb = w % Ntiles;
  v8f acc = {};
  for (int kc = 0; kc < KC; ++kc) {
    v16h a = *(const v16h*)(A + (((size_t)mb * KC + kc) * 32 + lane) * 16);
    v16h b = *(const v16h*)(B + (((size_t)nb * KC + kc) * 32 + lane) * 16);
    acc = wmma_f16(a, b, acc);
  }
  int Ncols = Ntiles * 16;
  int col = nb * 16 + nl;
  float bv = bias ? bias[col] : 0.0f;
#pragma unroll
  for (int r = 0; r < 8; ++r) {
    int row = mb * 16 + r + 8 * hi;     // C layout: VGPR r, lanes 16-31 hold M=r+8
    float v = acc[r] + bv;
    if (do_relu) v = fmaxf(v, 0.0f);
    if (residual) v += residual[(size_t)row * Ncols + col];
    Cout[(size_t)row * Ncols + col] = v;
  }
}

// ---------------------------------------------------------------------------
// Flash attention, one wave per 16 queries, 8 waves/block share K/V chunks.
// Qf: A-frags (f = qb*4 + kc), Kf: B^T-frags (f = keytile*4 + kc),
// Vf: B-frags kc-major (f = kb*8 + g).  K/V chunk kb = contiguous 8KB each.
// K/V chunks are double-buffered in LDS via async global->LDS copies when the
// toolchain exposes them (ASYNCcnt path), else read directly from L2.
// ---------------------------------------------------------------------------
__global__ void flash_attn(const _Float16* __restrict__ Qf,
                           const _Float16* __restrict__ Kf,
                           const _Float16* __restrict__ Vf,
                           float* __restrict__ ctx) {
  __shared__ _Float16 pstage[8 * 16 * 32];       // 1KB per wave, P staging
#if USE_ASYNC_LDS
  __shared__ _Float16 kstage[2][8 * 512];        // 8KB per buffer
  __shared__ _Float16 vstage[2][8 * 512];
#endif
  int tid  = threadIdx.x;
  int lane = tid & 31;
  int wv   = tid >> 5;
  int qb   = blockIdx.x * 8 + wv;
  int hi = lane >> 4, nl = lane & 15;
  _Float16* pw = pstage + wv * 512;

#if USE_ASYNC_LDS
  auto issue_chunk = [&](int kb, int buf) {
    const char* kg = (const char*)(Kf + (size_t)kb * 4096);
    const char* vg = (const char*)(Vf + (size_t)kb * 4096);
    char* kl = (char*)&kstage[buf][0];
    char* vl = (char*)&vstage[buf][0];
    int o0 = tid * 16, o1 = (tid + 256) * 16;    // 8192B / 256 thr = 2 x b128
    __builtin_amdgcn_global_load_async_to_lds_b128(
        (b128_t*)(kg + o0), (b128_t*)(kl + o0), 0, 0);
    __builtin_amdgcn_global_load_async_to_lds_b128(
        (b128_t*)(kg + o1), (b128_t*)(kl + o1), 0, 0);
    __builtin_amdgcn_global_load_async_to_lds_b128(
        (b128_t*)(vg + o0), (b128_t*)(vl + o0), 0, 0);
    __builtin_amdgcn_global_load_async_to_lds_b128(
        (b128_t*)(vg + o1), (b128_t*)(vl + o1), 0, 0);
  };
  issue_chunk(0, 0);
#endif

  v16h aq[4];
#pragma unroll
  for (int kc = 0; kc < 4; ++kc)
    aq[kc] = *(const v16h*)(Qf + (((size_t)qb * 4 + kc) * 32 + lane) * 16);

  v8f acc[8] = {};
  float mrow[8], lrow[8];
#pragma unroll
  for (int r = 0; r < 8; ++r) { mrow[r] = -3.0e38f; lrow[r] = 0.0f; }

  for (int kb = 0; kb < 128; ++kb) {
    const _Float16* kfr;
    const _Float16* vfr;
#if USE_ASYNC_LDS
    if (kb + 1 < 128) {
      issue_chunk(kb + 1, (kb + 1) & 1);
      __builtin_amdgcn_s_wait_asynccnt(4);   // in-order: chunk kb's 4 copies done
    } else {
      __builtin_amdgcn_s_wait_asynccnt(0);
    }
    __syncthreads();                          // all waves' copy shares landed
    kfr = &kstage[kb & 1][0];
    vfr = &vstage[kb & 1][0];
#else
    kfr = Kf + (size_t)kb * 4096;
    vfr = Vf + (size_t)kb * 4096;
    if (kb + 1 < 128)
      __builtin_prefetch(Kf + (size_t)(kb + 1) * 4096, 0, 1);
#endif
    // S = Q K^T for 32 keys (two 16x16 tiles)
    v8f s0 = {}, s1 = {};
#pragma unroll
    for (int kc = 0; kc < 4; ++kc) {
      v16h b0 = *(const v16h*)(kfr + ((size_t)(0 * 4 + kc) * 32 + lane) * 16);
      v16h b1 = *(const v16h*)(kfr + ((size_t)(1 * 4 + kc) * 32 + lane) * 16);
      s0 = wmma_f16(aq[kc], b0, s0);
      s1 = wmma_f16(aq[kc], b1, s1);
    }
    // online softmax per row (row m = r + 8*hi, 16 lanes per row)
#pragma unroll
    for (int r = 0; r < 8; ++r) {
      float mx = fmaxf(s0[r], s1[r]);
      for (int off = 8; off >= 1; off >>= 1) mx = fmaxf(mx, __shfl_xor(mx, off, 32));
      float nm = fmaxf(mrow[r], mx);
      float scale = __expf(mrow[r] - nm);
      float p0 = __expf(s0[r] - nm);
      float p1 = __expf(s1[r] - nm);
      float ps = p0 + p1;
      for (int off = 8; off >= 1; off >>= 1) ps += __shfl_xor(ps, off, 32);
      lrow[r] = lrow[r] * scale + ps;
      mrow[r] = nm;
#pragma unroll
      for (int g = 0; g < 8; ++g) acc[g][r] *= scale;
      int m = r + 8 * hi;
      pw[m * 32 + nl]      = (_Float16)p0;   // C-layout -> row-major P in LDS
      pw[m * 32 + 16 + nl] = (_Float16)p1;
    }
    // reload P as A-fragment (LDS in-order within wave)
    v16h ap;
#pragma unroll
    for (int h = 0; h < 16; ++h) ap[h] = pw[nl * 32 + kidx_of(h, hi)];
    // ctx += P V
#pragma unroll
    for (int g = 0; g < 8; ++g) {
      v16h bv = *(const v16h*)(vfr + ((size_t)g * 32 + lane) * 16);
      acc[g] = wmma_f16(ap, bv, acc[g]);
    }
#if USE_ASYNC_LDS
    __syncthreads();                          // buffer reads done before reuse
#endif
  }
#pragma unroll
  for (int g = 0; g < 8; ++g)
#pragma unroll
    for (int r = 0; r < 8; ++r) {
      int row = qb * 16 + r + 8 * hi;
      ctx[(size_t)row * DIM + g * 16 + nl] = acc[g][r] / lrow[r];
    }
}

// ---------------------------------------------------------------------------
// x = layernorm(proj + x_in) ; proj already has bd from gemm bias.
// One wave per row of 128.
// ---------------------------------------------------------------------------
__global__ void res_layernorm(const float* __restrict__ proj, const float* __restrict__ xin,
                              const float* __restrict__ g, const float* __restrict__ b,
                              float* __restrict__ xout) {
  int row = blockIdx.x * 8 + (threadIdx.x >> 5);
  int lane = threadIdx.x & 31;
  const float* pr = proj + (size_t)row * DIM;
  const float* xr = xin + (size_t)row * DIM;
  float v[4];
  float s = 0.0f;
#pragma unroll
  for (int i = 0; i < 4; ++i) { v[i] = pr[lane + 32 * i] + xr[lane + 32 * i]; s += v[i]; }
  for (int off = 16; off >= 1; off >>= 1) s += __shfl_xor(s, off, 32);
  float mean = s * (1.0f / 128.0f);
  float vs = 0.0f;
#pragma unroll
  for (int i = 0; i < 4; ++i) { float d = v[i] - mean; vs += d * d; }
  for (int off = 16; off >= 1; off >>= 1) vs += __shfl_xor(vs, off, 32);
  float rstd = rsqrtf(vs * (1.0f / 128.0f) + 1e-5f);
#pragma unroll
  for (int i = 0; i < 4; ++i) {
    int c = lane + 32 * i;
    xout[(size_t)row * DIM + c] = (v[i] - mean) * rstd * g[c] + b[c];
  }
}

__global__ void build_feat(const int* __restrict__ idx, const float* __restrict__ x,
                           float* __restrict__ feat) {
  int t = blockIdx.x * blockDim.x + threadIdx.x;
  if (t >= NEDGES * 256) return;
  int e = t >> 8, c = t & 255;
  int node = (c < 128) ? idx[2 * e] : idx[2 * e + 1];
  feat[t] = x[(size_t)node * DIM + (c & 127)];
}

__global__ void zero2(float* __restrict__ a, float* __restrict__ b, int n) {
  int t = blockIdx.x * blockDim.x + threadIdx.x;
  if (t < n) { a[t] = 0.0f; b[t] = 0.0f; }
}

// 256->4 head, softmax, binary marginals + atomic segment sums
__global__ void edge_out(const float* __restrict__ h, const float* __restrict__ wf,
                         const float* __restrict__ bf, const int* __restrict__ idx,
                         float* __restrict__ out, float* __restrict__ sums,
                         float* __restrict__ cnts) {
  int e = blockIdx.x * blockDim.x + threadIdx.x;
  if (e >= NEDGES) return;
  float lg[4] = { bf[0], bf[1], bf[2], bf[3] };
  const float* hr = h + (size_t)e * 256;
  for (int k = 0; k < 256; ++k) {
    float hv = hr[k];
#pragma unroll
    for (int c = 0; c < 4; ++c) lg[c] += hv * wf[c * 256 + k];
  }
  float mx = fmaxf(fmaxf(lg[0], lg[1]), fmaxf(lg[2], lg[3]));
  float p[4], sum = 0.0f;
#pragma unroll
  for (int c = 0; c < 4; ++c) { p[c] = __expf(lg[c] - mx); sum += p[c]; }
  float inv = 1.0f / sum;
#pragma unroll
  for (int c = 0; c < 4; ++c) { p[c] *= inv; out[4096 + e * 4 + c] = p[c]; }
  float pi = p[2] + p[3];   // marginals[:,1,:].sum(-1)
  float pj = p[1] + p[3];   // marginals[:,:,1].sum(-1)
  int i = idx[2 * e], j = idx[2 * e + 1];
  atomicAdd(&sums[i], pi); atomicAdd(&cnts[i], 1.0f);
  atomicAdd(&sums[j], pj); atomicAdd(&cnts[j], 1.0f);
}

__global__ void finalize_unary(const float* __restrict__ sums, const float* __restrict__ cnts,
                               float* __restrict__ out) {
  int t = blockIdx.x * blockDim.x + threadIdx.x;
  if (t < NNODES) out[t] = sums[t] / cnts[t];
}

// ---------------------------------------------------------------------------
extern "C" void kernel_launch(void* const* d_in, const int* in_sizes, int n_in,
                              void* d_out, int out_size, void* d_ws, size_t ws_size,
                              hipStream_t stream) {
  (void)in_sizes; (void)n_in; (void)out_size; (void)ws_size;
  const int*   binary_idx = (const int*)d_in[0];
  const float* node_emb = (const float*)d_in[1];
  const float* Wq = (const float*)d_in[2];
  const float* bq = (const float*)d_in[3];
  const float* Wk = (const float*)d_in[4];
  const float* bk = (const float*)d_in[5];
  const float* Wv = (const float*)d_in[6];
  const float* bv = (const float*)d_in[7];
  const float* Wd = (const float*)d_in[8];
  const float* bd = (const float*)d_in[9];
  const float* ln_g = (const float*)d_in[10];
  const float* ln_b = (const float*)d_in[11];
  const float* r1_w1 = (const float*)d_in[12];
  const float* r1_b1 = (const float*)d_in[13];
  const float* r1_w2 = (const float*)d_in[14];
  const float* r1_b2 = (const float*)d_in[15];
  const float* r2_w1 = (const float*)d_in[16];
  const float* r2_b1 = (const float*)d_in[17];
  const float* r2_w2 = (const float*)d_in[18];
  const float* r2_b2 = (const float*)d_in[19];
  const float* wf = (const float*)d_in[20];
  const float* bf = (const float*)d_in[21];
  float* out = (float*)d_out;

  char* base = (char*)d_ws; size_t off = 0;
  auto alloc = [&](size_t bytes) -> void* {
    void* p = base + off; off = (off + bytes + 255) & ~(size_t)255; return p;
  };
  float* xb0  = (float*)alloc((size_t)NNODES * DIM * 4);
  float* xb1  = (float*)alloc((size_t)NNODES * DIM * 4);
  float* qbuf = (float*)alloc((size_t)NNODES * DIM * 4);
  float* kbuf = (float*)alloc((size_t)NNODES * DIM * 4);
  float* vbuf = (float*)alloc((size_t)NNODES * DIM * 4);
  float* ctx  = (float*)alloc((size_t)NNODES * DIM * 4);
  float* proj = (float*)alloc((size_t)NNODES * DIM * 4);
  _Float16* xA = (_Float16*)alloc((size_t)NNODES * DIM * 2);
  _Float16* qA = (_Float16*)alloc((size_t)NNODES * DIM * 2);
  _Float16* kB = (_Float16*)alloc((size_t)NNODES * DIM * 2);
  _Float16* vB = (_Float16*)alloc((size_t)NNODES * DIM * 2);
  _Float16* cA = (_Float16*)alloc((size_t)NNODES * DIM * 2);
  _Float16* wqB = (_Float16*)alloc((size_t)DIM * DIM * 2);
  _Float16* wkB = (_Float16*)alloc((size_t)DIM * DIM * 2);
  _Float16* wvB = (_Float16*)alloc((size_t)DIM * DIM * 2);
  _Float16* wdB = (_Float16*)alloc((size_t)DIM * DIM * 2);
  float* feat = (float*)alloc((size_t)NEDGES * 256 * 4);
  float* h1   = (float*)alloc((size_t)NEDGES * 256 * 4);
  float* h2   = (float*)alloc((size_t)NEDGES * 256 * 4);
  _Float16* fA = (_Float16*)alloc((size_t)NEDGES * 256 * 2);
  _Float16* m1B = (_Float16*)alloc((size_t)256 * 256 * 2);
  _Float16* m2B = (_Float16*)alloc((size_t)256 * 256 * 2);
  _Float16* m3B = (_Float16*)alloc((size_t)256 * 256 * 2);
  _Float16* m4B = (_Float16*)alloc((size_t)256 * 256 * 2);
  float* sums = (float*)alloc((size_t)NNODES * 4);
  float* cnts = (float*)alloc((size_t)NNODES * 4);

  const int TPB = 256;
  auto nb = [](long threads) { return (int)((threads + 255) / 256); };

  const float* xin = node_emb;
  float* xouts[2] = { xb0, xb1 };
  for (int l = 0; l < 2; ++l) {
    const float* WqL = Wq + (size_t)l * DIM * DIM;
    const float* WkL = Wk + (size_t)l * DIM * DIM;
    const float* WvL = Wv + (size_t)l * DIM * DIM;
    const float* WdL = Wd + (size_t)l * DIM * DIM;
    // weight B^T fragments: 8 n-tiles x 4 k-chunks = 32 frags each
    pack_a_f32<<<nb(32 * 32), TPB, 0, stream>>>(WqL, wqB, 4, 32);
    pack_a_f32<<<nb(32 * 32), TPB, 0, stream>>>(WkL, wkB, 4, 32);
    pack_a_f32<<<nb(32 * 32), TPB, 0, stream>>>(WvL, wvB, 4, 32);
    pack_a_f32<<<nb(32 * 32), TPB, 0, stream>>>(WdL, wdB, 4, 32);
    // x A-fragments: 256 m-tiles x 4 k-chunks
    pack_a_f32<<<nb(1024L * 32), TPB, 0, stream>>>(xin, xA, 4, 1024);
    // q/k/v projections
    gemm16<<<nb(2048L * 32), TPB, 0, stream>>>(xA, wqB, bq + l * DIM, nullptr, qbuf, 256, 8, 4, 0);
    gemm16<<<nb(2048L * 32), TPB, 0, stream>>>(xA, wkB, bk + l * DIM, nullptr, kbuf, 256, 8, 4, 0);
    gemm16<<<nb(2048L * 32), TPB, 0, stream>>>(xA, wvB, bv + l * DIM, nullptr, vbuf, 256, 8, 4, 0);
    // attention fragments
    pack_a_f32<<<nb(1024L * 32), TPB, 0, stream>>>(qbuf, qA, 4, 1024);   // Q A-frags
    pack_a_f32<<<nb(1024L * 32), TPB, 0, stream>>>(kbuf, kB, 4, 1024);   // K^T B-frags
    pack_bN_f32<<<nb(1024L * 32), TPB, 0, stream>>>(vbuf, vB, 8, DIM, 1024); // V B-frags
    flash_attn<<<32, TPB, 0, stream>>>(qA, kB, vB, ctx);
    // output projection + residual + layernorm
    pack_a_f32<<<nb(1024L * 32), TPB, 0, stream>>>(ctx, cA, 4, 1024);
    gemm16<<<nb(2048L * 32), TPB, 0, stream>>>(cA, wdB, bd + l * DIM, nullptr, proj, 256, 8, 4, 0);
    res_layernorm<<<512, TPB, 0, stream>>>(proj, xin, ln_g + l * DIM, ln_b + l * DIM, xouts[l]);
    xin = xouts[l];
  }

  // edge head: feat = [x[i], x[j]]  (8064 x 256), two residual MLPs via WMMA
  build_feat<<<nb((long)NEDGES * 256), TPB, 0, stream>>>(binary_idx, xouts[1], feat);
  pack_a_f32<<<nb(128 * 32), TPB, 0, stream>>>(r1_w1, m1B, 8, 128);
  pack_a_f32<<<nb(128 * 32), TPB, 0, stream>>>(r1_w2, m2B, 8, 128);
  pack_a_f32<<<nb(128 * 32), TPB, 0, stream>>>(r2_w1, m3B, 8, 128);
  pack_a_f32<<<nb(128 * 32), TPB, 0, stream>>>(r2_w2, m4B, 8, 128);

  pack_a_f32<<<nb(4032L * 32), TPB, 0, stream>>>(feat, fA, 8, 4032);
  gemm16<<<nb(8064L * 32), TPB, 0, stream>>>(fA, m1B, r1_b1, nullptr, h1, 504, 16, 8, 1);
  pack_a_f32<<<nb(4032L * 32), TPB, 0, stream>>>(h1, fA, 8, 4032);
  gemm16<<<nb(8064L * 32), TPB, 0, stream>>>(fA, m2B, r1_b2, feat, h2, 504, 16, 8, 1);
  pack_a_f32<<<nb(4032L * 32), TPB, 0, stream>>>(h2, fA, 8, 4032);
  gemm16<<<nb(8064L * 32), TPB, 0, stream>>>(fA, m3B, r2_b1, nullptr, h1, 504, 16, 8, 1);
  pack_a_f32<<<nb(4032L * 32), TPB, 0, stream>>>(h1, fA, 8, 4032);
  gemm16<<<nb(8064L * 32), TPB, 0, stream>>>(fA, m4B, r2_b2, h2, h1, 504, 16, 8, 1);

  zero2<<<nb(NNODES), TPB, 0, stream>>>(sums, cnts, NNODES);
  edge_out<<<nb(NEDGES), TPB, 0, stream>>>(h1, wf, bf, binary_idx, out, sums, cnts);
  finalize_unary<<<nb(NNODES), TPB, 0, stream>>>(sums, cnts, out);
}
